// GraphNeuralNetwork_58042188038559
// MI455X (gfx1250) — matrix operand order repaired and checked by reading, not compiled, not run on hardware
//
#include <hip/hip_runtime.h>
#include <math.h>
#include <stdint.h>

// ---------------------------------------------------------------------------
// CDNA5 (gfx1250) WMMA types
// ---------------------------------------------------------------------------
typedef __attribute__((ext_vector_type(16))) __bf16 v16bf;
typedef __attribute__((ext_vector_type(8)))  float  v8f;

union FragBF {
    v16bf        v;
    unsigned int d[8];
};

// fp32 -> bf16 bits, round-to-nearest-even
__device__ __forceinline__ unsigned short f2bf_bits(float f) {
    union { float f; unsigned u; } c; c.f = f;
    unsigned r = c.u + 0x7FFFu + ((c.u >> 16) & 1u);
    return (unsigned short)(r >> 16);
}

// ---------------------------------------------------------------------------
// Async global->LDS 16-byte copy (ASYNCcnt-tracked); signature probe-confirmed:
// (v4i addrspace(1)*, v4i addrspace(3)*, imm offset, imm cpol)
// ---------------------------------------------------------------------------
typedef int v4i_ __attribute__((vector_size(16)));
typedef __attribute__((address_space(1))) v4i_* gptr_v4i;
typedef __attribute__((address_space(3))) v4i_* lptr_v4i;

#if __has_builtin(__builtin_amdgcn_global_load_async_to_lds_b128)
#define ASYNC_BUILTIN 1
#else
#define ASYNC_BUILTIN 0
#endif

__device__ __forceinline__ void async_copy16(void* lds, const void* g) {
#if ASYNC_BUILTIN
    __builtin_amdgcn_global_load_async_to_lds_b128(
        (gptr_v4i)(uintptr_t)g,
        (lptr_v4i)(uintptr_t)(unsigned)(uintptr_t)lds,
        0, 0);
#else
    unsigned ldsoff = (unsigned)(uintptr_t)lds;
    unsigned long long gaddr = (unsigned long long)(uintptr_t)g;
    asm volatile("global_load_async_to_lds_b128 %0, %1, off"
                 :: "v"(ldsoff), "v"(gaddr) : "memory");
#endif
}

__device__ __forceinline__ void async_wait_le8() {
#if __has_builtin(__builtin_amdgcn_s_wait_asynccnt)
    __builtin_amdgcn_s_wait_asynccnt(8);
#else
    asm volatile("s_wait_asynccnt 0x8" ::: "memory");
#endif
}
__device__ __forceinline__ void async_wait0() {
#if __has_builtin(__builtin_amdgcn_s_wait_asynccnt)
    __builtin_amdgcn_s_wait_asynccnt(0);
#else
    asm volatile("s_wait_asynccnt 0x0" ::: "memory");
#endif
}

// ---------------------------------------------------------------------------
// fp32 -> packed bf16 conversion (n2 = elements/2)
// ---------------------------------------------------------------------------
__global__ void cvt_f32_bf16(const float* __restrict__ in,
                             unsigned int* __restrict__ out, size_t n2)
{
    size_t i  = (size_t)blockIdx.x * blockDim.x + threadIdx.x;
    size_t st = (size_t)gridDim.x * blockDim.x;
    for (; i < n2; i += st) {
        float2 q = ((const float2*)in)[i];
        out[i] = (unsigned)f2bf_bits(q.x) | ((unsigned)f2bf_bits(q.y) << 16);
    }
}

// ---------------------------------------------------------------------------
// fp32 [K,N] -> bf16 transposed [N,K] (one-time, LDS-tiled 32x32)
// K%32==0, N%32==0. 256 threads = 32x8.
// ---------------------------------------------------------------------------
__global__ __launch_bounds__(256)
void cvt_transpose_f32_bf16(const float* __restrict__ in,
                            unsigned short* __restrict__ out, int K, int N)
{
    __shared__ float t[32][33];
    const int k0 = blockIdx.y * 32;
    const int n0 = blockIdx.x * 32;
    const int tx = threadIdx.x & 31;
    const int ty = threadIdx.x >> 5;
    #pragma unroll
    for (int i = ty; i < 32; i += 8)
        t[i][tx] = in[(size_t)(k0 + i) * N + n0 + tx];
    __syncthreads();
    #pragma unroll
    for (int i = ty; i < 32; i += 8)
        out[(size_t)(n0 + i) * K + k0 + tx] = f2bf_bits(t[tx][i]);
}

// ---------------------------------------------------------------------------
// Universal NT GEMM, bf16 operands, double-buffered async LDS staging:
//   C[M,N] = epilogue(A[M,K] @ concat(B1,B2)[N,K]^T + bias)
// B1 rows are [KH] wide (k in [0,KH)), B2 rows are [K-KH] wide (k in [KH,K)).
// Single-matrix case: pass KH == K (B2 unused). All B operands are K-major.
// Tile 64x64, BK=64, 128 threads = 4 waves (2x2), each wave 32x32.
// M%64==0, N%64==0, K%64==0, KH%64==0.
// act: 0 = none, 1 = relu, 2 = sigmoid. Outputs: fp32 Cf and/or bf16 Cb.
// ---------------------------------------------------------------------------
__global__ __launch_bounds__(128)
void gemm_nt_bf16(const unsigned short* __restrict__ A,
                  const unsigned short* __restrict__ B1,
                  const unsigned short* __restrict__ B2, int KH,
                  const float* __restrict__ bias,
                  float* __restrict__ Cf, unsigned short* __restrict__ Cb,
                  int M, int N, int K, int act)
{
    constexpr int BM = 64, BN = 64, BK = 64;
    __shared__ unsigned short As[2][BM][BK];   // 2 x 8 KB
    __shared__ unsigned short Bs[2][BN][BK];   // 2 x 8 KB

    const int tid  = threadIdx.x;
    const int lane = tid & 31;
    const int wave = tid >> 5;
    const int wm   = (wave >> 1) * 32;
    const int wn   = (wave & 1) * 32;
    const int m0   = blockIdx.y * BM;
    const int n0   = blockIdx.x * BN;
    const int lm   = lane & 15;
    const int lh   = lane >> 4;
    const int KB2  = K - KH;   // B2 row width (0 when single matrix; branch never taken)

    // stage one 64x64 bf16 tile pair: 8 async b128 instructions per wave
    auto stage = [&](int k0, int buf) {
        #pragma unroll
        for (int p0 = 0; p0 < 4; ++p0) {
            int p = tid + p0 * 128;           // 512 x 16B for A tile
            int r = p >> 3;
            int c = (p & 7) * 8;
            async_copy16(&As[buf][r][c], A + (size_t)(m0 + r) * K + k0 + c);
        }
        #pragma unroll
        for (int p0 = 0; p0 < 4; ++p0) {
            int p = tid + p0 * 128;           // 512 x 16B for B tile
            int r = p >> 3;
            int c = (p & 7) * 8;
            const unsigned short* src = (k0 < KH)
                ? (B1 + (size_t)(n0 + r) * KH + k0 + c)
                : (B2 + (size_t)(n0 + r) * KB2 + (k0 - KH) + c);
            async_copy16(&Bs[buf][r][c], src);
        }
    };

    v8f acc[2][2] = {};

    stage(0, 0);
    int buf = 0;
    for (int k0 = 0; k0 < K; k0 += BK, buf ^= 1) {
        const bool more = (k0 + BK) < K;
        if (more) stage(k0 + BK, buf ^ 1);    // fill other buffer while we compute
        if (more) async_wait_le8();           // this wave's current-buffer copies done
        else      async_wait0();
        __syncthreads();                      // all waves' copies visible

        #pragma unroll
        for (int ks = 0; ks < 2; ++ks) {      // two K=32 sub-steps per 64-wide tile
            FragBF a[2], b[2];
            #pragma unroll
            for (int i = 0; i < 2; ++i) {
                int row = wm + i * 16 + lm;
                #pragma unroll
                for (int vv = 0; vv < 8; ++vv) {
                    // A 16x32 bf16 layout: VGPR0..3 K={0,2,4,6}+8*lh; VGPR4..7 K={16..22}+8*lh
                    int k = ((vv < 4) ? (2 * vv) : (8 + 2 * vv)) + lh * 8 + ks * 32;
                    a[i].d[vv] = *(const unsigned int*)&As[buf][row][k];
                }
            }
            #pragma unroll
            for (int j = 0; j < 2; ++j) {
                int nb = wn + j * 16 + lm;
                #pragma unroll
                for (int vv = 0; vv < 8; ++vv) {
                    // B 32x16 layout: same K order as A: VGPR0..3 K={0..6}+8*lh; VGPR4..7 K={16..22}+8*lh
                    int k = ((vv < 4) ? (2 * vv) : (8 + 2 * vv)) + lh * 8 + ks * 32;
                    b[j].d[vv] = *(const unsigned int*)&Bs[buf][nb][k];
                }
            }
            #pragma unroll
            for (int i = 0; i < 2; ++i)
                #pragma unroll
                for (int j = 0; j < 2; ++j)
                    acc[i][j] = __builtin_amdgcn_wmma_f32_16x16x32_bf16(
                        false, a[i].v, false, b[j].v, (short)0, acc[i][j], false, false);
        }
        __syncthreads();                      // reads done before buffer reuse
    }

    #pragma unroll
    for (int i = 0; i < 2; ++i) {
        int mrow = m0 + wm + i * 16 + lh * 8;
        #pragma unroll
        for (int j = 0; j < 2; ++j) {
            int col = n0 + wn + j * 16 + lm;
            float bv = bias ? bias[col] : 0.0f;
            #pragma unroll
            for (int r = 0; r < 8; ++r) {
                float x = acc[i][j][r] + bv;
                if (act == 1) x = fmaxf(x, 0.0f);
                else if (act == 2) x = 1.0f / (1.0f + __expf(-x));
                size_t idx = (size_t)(mrow + r) * N + col;
                if (Cf) Cf[idx] = x;
                if (Cb) Cb[idx] = f2bf_bits(x);
            }
        }
    }
}

// ---------------------------------------------------------------------------
// SpMM scatter: out[row[e], :] += val[e] * x[col[e], :]   (fp32, d % 4 == 0)
// ---------------------------------------------------------------------------
__global__ void spmm_scatter(const int* __restrict__ row, const int* __restrict__ col,
                             const float* __restrict__ val, const float* __restrict__ x,
                             float* __restrict__ out, int d)
{
    int e = blockIdx.x;
    int r = row[e];
    int c = col[e];
    float v = val[e];
    const float4* xs = (const float4*)(x + (size_t)c * d);
    float* o = out + (size_t)r * d;
    for (int j = threadIdx.x; j < (d >> 2); j += blockDim.x) {
        float4 q = xs[j];
        int b = j * 4;
        atomicAdd(o + b + 0, v * q.x);
        atomicAdd(o + b + 1, v * q.y);
        atomicAdd(o + b + 2, v * q.z);
        atomicAdd(o + b + 3, v * q.w);
    }
}

__global__ void zero_f32(float* __restrict__ p, size_t n)
{
    size_t i  = (size_t)blockIdx.x * blockDim.x + threadIdx.x;
    size_t st = (size_t)gridDim.x * blockDim.x;
    for (; i < n; i += st) p[i] = 0.0f;
}

// out = act(in + bias[col]); optional fp32 and/or bf16 destinations
__global__ void bias_act_dual(const float* __restrict__ in, const float* __restrict__ bias,
                              float* __restrict__ outF, unsigned short* __restrict__ outB,
                              size_t n, int d, int act)
{
    size_t i  = (size_t)blockIdx.x * blockDim.x + threadIdx.x;
    size_t st = (size_t)gridDim.x * blockDim.x;
    for (; i < n; i += st) {
        int col = (int)(i % (size_t)d);
        float x = in[i] + bias[col];
        if (act == 1) x = fmaxf(x, 0.0f);
        if (outF) outF[i] = x;
        if (outB) outB[i] = f2bf_bits(x);
    }
}

// ---------------------------------------------------------------------------
// Orchestration
// ---------------------------------------------------------------------------
extern "C" void kernel_launch(void* const* d_in, const int* in_sizes, int n_in,
                              void* d_out, int out_size, void* d_ws, size_t ws_size,
                              hipStream_t stream)
{
    (void)in_sizes; (void)n_in; (void)out_size; (void)ws_size;

    const float* seq    = (const float*)d_in[0];
    const float* go     = (const float*)d_in[1];
    const int*   erow   = (const int*)d_in[2];
    const int*   ecol   = (const int*)d_in[3];
    const float* evalv  = (const float*)d_in[4];
    const float* mlp_w1 = (const float*)d_in[5];
    const float* mlp_b1 = (const float*)d_in[6];
    const float* mlp_w2 = (const float*)d_in[7];
    const float* mlp_b2 = (const float*)d_in[8];
    const float* gc1_w  = (const float*)d_in[9];
    const float* gc1_b  = (const float*)d_in[10];
    const float* gc2_w  = (const float*)d_in[11];
    const float* gc2_b  = (const float*)d_in[12];
    const float* se_w1  = (const float*)d_in[13];
    const float* se_b1  = (const float*)d_in[14];
    const float* se_w2  = (const float*)d_in[15];
    const float* se_b2  = (const float*)d_in[16];

    const int NGO = 40000, GOF = 512, SEQF = 1280, H0 = 1024, H1 = 512;
    const int NE = 640000, B = 4096;

    float* out   = (float*)d_out;
    float* h_sem = out;
    float* h_str = out + (size_t)NGO * H1;
    float* pred  = out + (size_t)2 * NGO * H1;

    // ---- workspace carve-up, 256B aligned regions
    char*  base = (char*)d_ws;
    size_t off  = 0;
    auto alloc = [&](size_t bytes) {
        void* p = base + off;
        off += (bytes + 255) & ~(size_t)255;
        return p;
    };
    float*          f32A  = (float*)alloc((size_t)NGO * H0 * 4);   // s1 accum; later xw2+s2
    float*          f32B  = (float*)alloc((size_t)NGO * H0 * 4);   // xw1; aliases h1B/xB (bf16)
    unsigned short* goB   = (unsigned short*)alloc((size_t)NGO * GOF * 2);  // later h_strB
    unsigned short* hsemB = (unsigned short*)alloc((size_t)NGO * H1 * 2);
    unsigned short* w1t   = (unsigned short*)alloc((size_t)GOF * H0 * 2);   // [H0,GOF]
    unsigned short* w2t   = (unsigned short*)alloc((size_t)H0 * H1 * 2);    // [H1,H0]
    unsigned short* g1t   = (unsigned short*)alloc((size_t)GOF * H0 * 2);   // [H0,GOF]
    unsigned short* g2t   = (unsigned short*)alloc((size_t)H0 * H1 * 2);    // [H1,H0]
    unsigned short* s1t   = (unsigned short*)alloc((size_t)SEQF * H0 * 2);  // [H0,SEQF]
    unsigned short* s2t   = (unsigned short*)alloc((size_t)H0 * H0 * 2);    // [H0,H0]
    unsigned short* seqB  = (unsigned short*)alloc((size_t)B * SEQF * 2);
    unsigned short* shB   = (unsigned short*)alloc((size_t)B * H0 * 2);
    unsigned short* soutB = (unsigned short*)alloc((size_t)B * H0 * 2);

    unsigned short* h1B   = (unsigned short*)f32B;   // [40000,1024] bf16, alias
    unsigned short* xB    = (unsigned short*)f32B;   // [40000,1024] bf16, alias (xw1 dead)
    unsigned short* hstrB = goB;                     // [40000,512]  bf16, alias (goB dead)

    dim3 blk(128);

    // ---- one-time bf16 conversions; weights transposed to K-major [N,K]
    cvt_f32_bf16<<<1024, 256, 0, stream>>>(go,  (unsigned*)goB,  (size_t)NGO * GOF / 2);
    cvt_f32_bf16<<<1024, 256, 0, stream>>>(seq, (unsigned*)seqB, (size_t)B * SEQF / 2);
    cvt_transpose_f32_bf16<<<dim3(H0 / 32, GOF / 32), 256, 0, stream>>>(mlp_w1, w1t, GOF, H0);
    cvt_transpose_f32_bf16<<<dim3(H1 / 32, H0 / 32), 256, 0, stream>>>(mlp_w2, w2t, H0, H1);
    cvt_transpose_f32_bf16<<<dim3(H0 / 32, GOF / 32), 256, 0, stream>>>(gc1_w, g1t, GOF, H0);
    cvt_transpose_f32_bf16<<<dim3(H1 / 32, H0 / 32), 256, 0, stream>>>(gc2_w, g2t, H0, H1);
    cvt_transpose_f32_bf16<<<dim3(H0 / 32, SEQF / 32), 256, 0, stream>>>(se_w1, s1t, SEQF, H0);
    cvt_transpose_f32_bf16<<<dim3(H0 / 32, H0 / 32), 256, 0, stream>>>(se_w2, s2t, H0, H0);

    // ---- semantic branch
    gemm_nt_bf16<<<dim3(H0 / 64, NGO / 64), blk, 0, stream>>>(
        goB, w1t, w1t, GOF, mlp_b1, nullptr, h1B, NGO, H0, GOF, 1);
    gemm_nt_bf16<<<dim3(H1 / 64, NGO / 64), blk, 0, stream>>>(
        h1B, w2t, w2t, H0, mlp_b2, h_sem, hsemB, NGO, H1, H0, 0);

    // ---- structure branch, layer 1
    gemm_nt_bf16<<<dim3(H0 / 64, NGO / 64), blk, 0, stream>>>(
        goB, g1t, g1t, GOF, nullptr, f32B, nullptr, NGO, H0, GOF, 0);   // xw1 (h1B dead)
    zero_f32<<<2048, 256, 0, stream>>>(f32A, (size_t)NGO * H0);
    spmm_scatter<<<NE, 256, 0, stream>>>(erow, ecol, evalv, f32B, f32A, H0);
    bias_act_dual<<<2048, 256, 0, stream>>>(f32A, gc1_b, nullptr, xB,
                                            (size_t)NGO * H0, H0, 1);

    // ---- structure branch, layer 2
    gemm_nt_bf16<<<dim3(H1 / 64, NGO / 64), blk, 0, stream>>>(
        xB, g2t, g2t, H0, nullptr, f32A, nullptr, NGO, H1, H0, 0);      // xw2 -> f32A[0:20.48M]
    float* s2 = f32A + (size_t)NGO * H1;
    zero_f32<<<2048, 256, 0, stream>>>(s2, (size_t)NGO * H1);
    spmm_scatter<<<NE, 128, 0, stream>>>(erow, ecol, evalv, f32A, s2, H1);
    bias_act_dual<<<2048, 256, 0, stream>>>(s2, gc2_b, h_str, hstrB,
                                            (size_t)NGO * H1, H1, 1);

    // ---- sequence encoder
    gemm_nt_bf16<<<dim3(H0 / 64, B / 64), blk, 0, stream>>>(
        seqB, s1t, s1t, SEQF, se_b1, nullptr, shB, B, H0, SEQF, 1);
    gemm_nt_bf16<<<dim3((2 * H1) / 64, B / 64), blk, 0, stream>>>(
        shB, s2t, s2t, H0, se_b2, nullptr, soutB, B, 2 * H1, H0, 0);

    // ---- prediction: sigmoid(seq_out @ concat(h_sem,h_str)^T)
    gemm_nt_bf16<<<dim3(NGO / 64, B / 64), blk, 0, stream>>>(
        soutB, hsemB, hstrB, 512, nullptr, pred, nullptr, B, NGO, 1024, 2);
}
